// PointNet2Wrapper_22806276341869
// MI455X (gfx1250) — compile-verified, matrix operand order checked by reading, not run on hardware
//
#include <hip/hip_runtime.h>

// ---------------------------------------------------------------------------
// PointNet++ SSG forward for MI455X (gfx1250, wave32, WMMA).
// Pipeline per SA stage: FPS -> gather -> ball query -> group (f16) ->
//   [GEMM(WMMA f16, f32 acc, fused prev-layer BN+ReLU on A) -> channel stats
//    -> BN scale/shift] x3 -> BN+ReLU+maxpool.
// GEMM: block tile 128x64, 8 waves (4Mx2N), wave tile 32x32 ->
//   4 x v_wmma_f32_16x16x32_f16 per K-step, fragments via ds_load_b128.
// FUSED template: K%32==0, branch-free vector staging with BN+ReLU on A.
// non-FUSED: ragged-K first layers (K=4/131/259); branchless staging via
//   clamped-index unconditional loads + v_cndmask zero-fill.
// ---------------------------------------------------------------------------

typedef __attribute__((ext_vector_type(16))) _Float16 v16h;
typedef __attribute__((ext_vector_type(8)))  _Float16 v8h;
typedef __attribute__((ext_vector_type(8)))  float    v8f;

union AFrag { v16h v; v8h h[2]; };
union CFrag { v8f v; float f[8]; };
union H8    { v8h v; _Float16 e[8]; };
union F4    { float4 v; float f[4]; };

// ------------------------------ FPS ----------------------------------------
__global__ __launch_bounds__(512)
void fps_kernel(const float* __restrict__ xyz, int N, int stride, int npoint,
                int* __restrict__ out_idx)
{
    const int b   = blockIdx.x;
    const int tid = threadIdx.x;
    const int NT  = 512;
    const float* base = xyz + (size_t)b * N * stride;

    float ld[16];
    const int cnt = (N + NT - 1) / NT;
    for (int j = 0; j < cnt; ++j) ld[j] = 1e10f;

    __shared__ float sval[512];
    __shared__ int   sidx[512];
    __shared__ int   sfar;

    int farthest = 0;
    for (int it = 0; it < npoint; ++it) {
        if (tid == 0) out_idx[b * npoint + it] = farthest;
        const float cx = base[farthest * stride + 0];
        const float cy = base[farthest * stride + 1];
        const float cz = base[farthest * stride + 2];

        float bv = -1.0f; int bi = 0x7fffffff;
        for (int j = 0; j < cnt; ++j) {
            const int i = tid + j * NT;
            if (i < N) {
                const float dx = base[i * stride + 0] - cx;
                const float dy = base[i * stride + 1] - cy;
                const float dz = base[i * stride + 2] - cz;
                float d = dx * dx + dy * dy + dz * dz;
                d = fminf(ld[j], d);
                ld[j] = d;
                if (d > bv || (d == bv && i < bi)) { bv = d; bi = i; }
            }
        }
        sval[tid] = bv; sidx[tid] = bi;
        __syncthreads();
        for (int off = 256; off > 0; off >>= 1) {
            if (tid < off) {
                const float v2 = sval[tid + off]; const int i2 = sidx[tid + off];
                if (v2 > sval[tid] || (v2 == sval[tid] && i2 < sidx[tid])) {
                    sval[tid] = v2; sidx[tid] = i2;
                }
            }
            __syncthreads();
        }
        if (tid == 0) sfar = sidx[0];
        __syncthreads();
        farthest = sfar;
    }
}

// ------------------------------ gather -------------------------------------
__global__ void gather_xyz_kernel(const float* __restrict__ xyz, int N, int stride,
                                  const int* __restrict__ idx, int S, int total,
                                  float* __restrict__ out)
{
    const int t = blockIdx.x * blockDim.x + threadIdx.x;
    if (t >= total) return;
    const int b = t / S;
    const int id = idx[t];
    const float* p = xyz + ((size_t)b * N + id) * stride;
    out[(size_t)t * 3 + 0] = p[0];
    out[(size_t)t * 3 + 1] = p[1];
    out[(size_t)t * 3 + 2] = p[2];
}

// ---------------------------- ball query -----------------------------------
__global__ __launch_bounds__(256)
void ball_query_kernel(const float* __restrict__ xyz, int N, int stride,
                       const float* __restrict__ new_xyz, int S,
                       float r2, int nsample, int* __restrict__ out)
{
    const int lane = threadIdx.x & 31;
    const int wq   = (blockIdx.x * blockDim.x + threadIdx.x) >> 5;
    const int b = wq / S, s = wq % S;

    const float* c = new_xyz + (size_t)(b * S + s) * 3;
    const float cx = c[0], cy = c[1], cz = c[2];
    int* o = out + (size_t)(b * S + s) * nsample;

    int count = 0;
    int first_i = 0;
    for (int basei = 0; basei < N; basei += 32) {
        const int i = basei + lane;
        bool pred = false;
        if (i < N) {
            const float* p = xyz + ((size_t)b * N + i) * stride;
            const float dx = p[0] - cx, dy = p[1] - cy, dz = p[2] - cz;
            pred = (dx * dx + dy * dy + dz * dz <= r2);
        }
        const unsigned mask = (unsigned)__ballot(pred);
        if (count == 0 && mask) first_i = basei + (__ffs(mask) - 1);
        const int prefix = __popc(mask & ((1u << lane) - 1u));
        if (pred) {
            const int slot = count + prefix;
            if (slot < nsample) o[slot] = i;
        }
        count += __popc(mask);
        if (count >= nsample) break;
    }
    const int start = count < nsample ? count : nsample;
    for (int j = start + lane; j < nsample; j += 32) o[j] = first_i;
}

// ------------------------------ grouping -----------------------------------
__global__ void group1_kernel(const float* __restrict__ points,
                              const int* __restrict__ ballidx,
                              const float* __restrict__ new_xyz,
                              _Float16* __restrict__ A1,
                              int total, int S, int ns, int N)
{
    const int t = blockIdx.x * blockDim.x + threadIdx.x;
    if (t >= total) return;
    const int bs = t / ns;
    const int b  = bs / S;
    const int id = ballidx[t];
    const float* p = points + ((size_t)b * N + id) * 4;
    const float* c = new_xyz + (size_t)bs * 3;
    _Float16* o = A1 + (size_t)t * 4;
    o[0] = (_Float16)(p[0] - c[0]);
    o[1] = (_Float16)(p[1] - c[1]);
    o[2] = (_Float16)(p[2] - c[2]);
    o[3] = (_Float16)p[3];
}

__global__ void group2_kernel(const float* __restrict__ l1_xyz,
                              const float* __restrict__ l1_pts,
                              const float* __restrict__ l2_xyz,
                              const int* __restrict__ ballidx,
                              _Float16* __restrict__ A2, int total)
{
    const int t = blockIdx.x * blockDim.x + threadIdx.x;
    if (t >= total) return;
    const int Cin = 131, S1 = 512, S2 = 128, ns = 64, Cp = 128;
    const int row = t / Cin, c = t % Cin;
    const int b = row / (S2 * ns);
    const int s = (row / ns) % S2;
    const int id = ballidx[row];
    float v;
    if (c < 3)
        v = l1_xyz[((size_t)b * S1 + id) * 3 + c] - l2_xyz[((size_t)b * S2 + s) * 3 + c];
    else
        v = l1_pts[((size_t)b * S1 + id) * Cp + (c - 3)];
    A2[t] = (_Float16)v;
}

__global__ void group3_kernel(const float* __restrict__ l2_xyz,
                              const float* __restrict__ l2_pts,
                              _Float16* __restrict__ A3, int total)
{
    const int t = blockIdx.x * blockDim.x + threadIdx.x;
    if (t >= total) return;
    const int Cin = 259;
    const int row = t / Cin, c = t % Cin;
    float v;
    if (c < 3) v = l2_xyz[(size_t)row * 3 + c];
    else       v = l2_pts[(size_t)row * 256 + (c - 3)];
    A3[t] = (_Float16)v;
}

// --------------------------- WMMA GEMM -------------------------------------
// Y[M,N] = act(A)[M,K] x W^T[K,N] + bias.
// FUSED=true : K%32==0, A staged with vector loads + BN(scale,shift)+ReLU.
// FUSED=false: ragged K (zero-padded to 32), raw A (first layer of a stage);
//              branchless: clamped-index unconditional loads + select-zero.
template <bool FUSED>
__global__ __launch_bounds__(256)
void gemm_wmma_kernel(const _Float16* __restrict__ A, int M, int K,
                      const float* __restrict__ in_scale,
                      const float* __restrict__ in_shift,
                      const float* __restrict__ W,
                      const float* __restrict__ bias, int N,
                      _Float16* __restrict__ Y)
{
    __shared__ __align__(16) _Float16 sA[128][40];  // 128 rows x 32 K (+pad)
    __shared__ __align__(16) _Float16 sB[64][40];   // 64 n-rows x 32 K (+pad)

    const int tid  = threadIdx.x;
    const int lane = tid & 31, wid = tid >> 5;
    const int wm = wid >> 1, wn = wid & 1;     // 4 x 2 waves
    const int m0 = blockIdx.x * 128;
    const int n0 = blockIdx.y * 64;
    const int hi = lane >> 4, lm = lane & 15;

    v8f acc[2][2];
#pragma unroll
    for (int i = 0; i < 2; ++i)
#pragma unroll
        for (int j = 0; j < 2; ++j)
            acc[i][j] = (v8f){0.f,0.f,0.f,0.f,0.f,0.f,0.f,0.f};

    for (int kk = 0; kk < K; kk += 32) {
        // ---- stage A tile: 128 rows x 32 K ; 16 halfs per thread ----
        {
            const int row  = tid >> 1;
            const int colb = (tid & 1) * 16;
            const _Float16* arow = A + (size_t)(m0 + row) * K;  // row base
            if (FUSED) {
                const _Float16* ap = arow + kk + colb;
                H8 h0, h1, o0, o1;
                h0.v = *reinterpret_cast<const v8h*>(ap);
                h1.v = *reinterpret_cast<const v8h*>(ap + 8);
                F4 s0, s1, s2, s3, t0, t1, t2, t3;
                const float* sp = in_scale + kk + colb;
                const float* tp = in_shift + kk + colb;
                s0.v = *(const float4*)(sp);      s1.v = *(const float4*)(sp + 4);
                s2.v = *(const float4*)(sp + 8);  s3.v = *(const float4*)(sp + 12);
                t0.v = *(const float4*)(tp);      t1.v = *(const float4*)(tp + 4);
                t2.v = *(const float4*)(tp + 8);  t3.v = *(const float4*)(tp + 12);
#pragma unroll
                for (int j = 0; j < 4; ++j) {
                    o0.e[j]     = (_Float16)fmaxf((float)h0.e[j]     * s0.f[j] + t0.f[j], 0.f);
                    o0.e[j + 4] = (_Float16)fmaxf((float)h0.e[j + 4] * s1.f[j] + t1.f[j], 0.f);
                    o1.e[j]     = (_Float16)fmaxf((float)h1.e[j]     * s2.f[j] + t2.f[j], 0.f);
                    o1.e[j + 4] = (_Float16)fmaxf((float)h1.e[j + 4] * s3.f[j] + t3.f[j], 0.f);
                }
                *reinterpret_cast<v8h*>(&sA[row][colb])     = o0.v;
                *reinterpret_cast<v8h*>(&sA[row][colb + 8]) = o1.v;
                if (kk + 32 < K) __builtin_prefetch(ap + 32, 0, 1);
            } else {
                // Branchless ragged staging: clamp index into [0,K), load
                // unconditionally (in-bounds), then select zero for pad lanes.
                H8 o0, o1;
#pragma unroll
                for (int j = 0; j < 8; ++j) {
                    const int k0 = kk + colb + j;
                    const int k1 = k0 + 8;
                    const int kc0 = (k0 < K) ? k0 : 0;
                    const int kc1 = (k1 < K) ? k1 : 0;
                    const _Float16 r0 = arow[kc0];
                    const _Float16 r1 = arow[kc1];
                    o0.e[j] = (k0 < K) ? r0 : (_Float16)0.f;
                    o1.e[j] = (k1 < K) ? r1 : (_Float16)0.f;
                }
                *reinterpret_cast<v8h*>(&sA[row][colb])     = o0.v;
                *reinterpret_cast<v8h*>(&sA[row][colb + 8]) = o1.v;
            }
        }
        // ---- stage B tile: sB[n][k] = W[n0+n][kk+k] (f32 -> f16) ----
        {
            const int n    = tid >> 2;
            const int colb = (tid & 3) * 8;
            const float* wrow = W + (size_t)(n0 + n) * K;  // row base
            H8 o;
            if (FUSED) {
                const float* wp = wrow + kk + colb;
                F4 w0, w1;
                w0.v = *(const float4*)(wp);
                w1.v = *(const float4*)(wp + 4);
#pragma unroll
                for (int j = 0; j < 4; ++j) {
                    o.e[j]     = (_Float16)w0.f[j];
                    o.e[j + 4] = (_Float16)w1.f[j];
                }
            } else {
#pragma unroll
                for (int j = 0; j < 8; ++j) {
                    const int k  = kk + colb + j;
                    const int kc = (k < K) ? k : 0;
                    const float r = wrow[kc];
                    o.e[j] = (k < K) ? (_Float16)r : (_Float16)0.f;
                }
            }
            *reinterpret_cast<v8h*>(&sB[n][colb]) = o.v;
        }
        __syncthreads();

        // ---- fragments (ISA 7.12.2): lane<16 -> K 0-7,16-23 ; lane>=16 ->
        //      K 8-15,24-31 ; each frag = two ds_load_b128 (row stride 80B) ----
        AFrag a[2], bf[2];
#pragma unroll
        for (int i = 0; i < 2; ++i) {
            const int ar = wm * 32 + i * 16 + lm;
            a[i].h[0] = *reinterpret_cast<const v8h*>(&sA[ar][hi * 8]);
            a[i].h[1] = *reinterpret_cast<const v8h*>(&sA[ar][16 + hi * 8]);
        }
#pragma unroll
        for (int j = 0; j < 2; ++j) {
            const int br = wn * 32 + j * 16 + lm;
            bf[j].h[0] = *reinterpret_cast<const v8h*>(&sB[br][hi * 8]);
            bf[j].h[1] = *reinterpret_cast<const v8h*>(&sB[br][16 + hi * 8]);
        }
#pragma unroll
        for (int i = 0; i < 2; ++i)
#pragma unroll
            for (int j = 0; j < 2; ++j)
                acc[i][j] = __builtin_amdgcn_wmma_f32_16x16x32_f16(
                    false, a[i].v, false, bf[j].v, (short)0, acc[i][j], false, false);
        __syncthreads();
    }

    // ---- epilogue: +bias, store pre-BN f16 (C/D: lane -> n, vgpr -> m) ----
#pragma unroll
    for (int j = 0; j < 2; ++j) {
        const int n = n0 + wn * 32 + j * 16 + lm;
        const float bn_ = bias[n];
#pragma unroll
        for (int i = 0; i < 2; ++i) {
            CFrag u; u.v = acc[i][j];
#pragma unroll
            for (int v = 0; v < 8; ++v) {
                const int m = m0 + wm * 32 + i * 16 + v + 8 * hi;
                Y[(size_t)m * N + n] = (_Float16)(u.f[v] + bn_);
            }
        }
    }
}

// ---------------------- channel stats + BN finalize ------------------------
__global__ __launch_bounds__(256)
void channel_stats_kernel(const _Float16* __restrict__ Y, int M, int N,
                          float* __restrict__ sum, float* __restrict__ sumsq)
{
    const int n = blockIdx.x;
    float s = 0.f, q = 0.f;
    for (int m = threadIdx.x; m < M; m += 256) {
        const float v = (float)Y[(size_t)m * N + n];
        s += v; q += v * v;
    }
    __shared__ float ss[256], sq[256];
    ss[threadIdx.x] = s; sq[threadIdx.x] = q;
    __syncthreads();
    for (int off = 128; off > 0; off >>= 1) {
        if (threadIdx.x < off) {
            ss[threadIdx.x] += ss[threadIdx.x + off];
            sq[threadIdx.x] += sq[threadIdx.x + off];
        }
        __syncthreads();
    }
    if (threadIdx.x == 0) { sum[n] = ss[0]; sumsq[n] = sq[0]; }
}

__global__ void bn_finalize_kernel(const float* __restrict__ sum,
                                   const float* __restrict__ sumsq,
                                   const float* __restrict__ g,
                                   const float* __restrict__ beta,
                                   float invM, int N,
                                   float* __restrict__ scale,
                                   float* __restrict__ shift)
{
    const int n = blockIdx.x * blockDim.x + threadIdx.x;
    if (n >= N) return;
    const float mean = sum[n] * invM;
    const float var  = sumsq[n] * invM - mean * mean;
    const float sc   = g[n] * rsqrtf(var + 1e-5f);
    scale[n] = sc;
    shift[n] = beta[n] - mean * sc;
}

// ------------------------- BN + ReLU + max-pool ----------------------------
__global__ void bn_relu_maxpool_kernel(const _Float16* __restrict__ Y,
                                       const float* __restrict__ scale,
                                       const float* __restrict__ shift,
                                       int R, int Kn, int C,
                                       float* __restrict__ out)
{
    const int t = blockIdx.x * blockDim.x + threadIdx.x;
    if (t >= R * C) return;
    const int r = t / C, c = t % C;
    const float sc = scale[c], sh = shift[c];
    const size_t base = (size_t)r * Kn * C + c;
    float m = -1e30f;
    for (int j = 0; j < Kn; ++j) {
        const float v = (float)Y[base + (size_t)j * C] * sc + sh;
        m = fmaxf(m, v);
    }
    out[t] = fmaxf(m, 0.0f);  // max of relu == relu of max
}

// ===========================================================================
extern "C" void kernel_launch(void* const* d_in, const int* in_sizes, int n_in,
                              void* d_out, int out_size, void* d_ws, size_t ws_size,
                              hipStream_t stream)
{
    (void)in_sizes; (void)n_in; (void)out_size; (void)ws_size;

    const float* points = (const float*)d_in[0];
    const float *Wp[9], *bp[9], *gp[9], *betap[9];
    for (int i = 0; i < 9; ++i) {
        Wp[i]    = (const float*)d_in[1 + 4 * i];
        bp[i]    = (const float*)d_in[2 + 4 * i];
        gp[i]    = (const float*)d_in[3 + 4 * i];
        betap[i] = (const float*)d_in[4 + 4 * i];
    }
    const int cins[9]  = {4, 64, 64, 131, 128, 128, 259, 256, 512};
    const int couts[9] = {64, 64, 128, 128, 128, 256, 256, 512, 1024};

    const int B = 16, N1 = 8192, S1 = 512, NS1 = 32, S2 = 128, NS2 = 64, K3 = 128;
    const int M1 = B * S1 * NS1;   // 262144
    const int M2 = B * S2 * NS2;   // 131072
    const int M3 = B * K3;         // 2048

    // ---------------- workspace carve-up ----------------
    char* ws = (char*)d_ws;
    size_t off = 0;
    auto alloc = [&](size_t bytes) -> char* {
        char* p = ws + off;
        off = (off + bytes + 255) & ~(size_t)255;
        return p;
    };
    int*      idx1    = (int*)alloc((size_t)B * S1 * 4);
    float*    l1_xyz  = (float*)alloc((size_t)B * S1 * 3 * 4);
    int*      ball1   = (int*)alloc((size_t)B * S1 * NS1 * 4);
    _Float16* A1      = (_Float16*)alloc((size_t)M1 * 4 * 2);
    int*      idx2    = (int*)alloc((size_t)B * S2 * 4);
    float*    l2_xyz  = (float*)alloc((size_t)B * S2 * 3 * 4);
    int*      ball2   = (int*)alloc((size_t)B * S2 * NS2 * 4);
    _Float16* A2      = (_Float16*)alloc((size_t)M2 * 131 * 2);
    _Float16* A3      = (_Float16*)alloc((size_t)M3 * 259 * 2);
    float*    l1_pts  = (float*)alloc((size_t)B * S1 * 128 * 4);
    float*    l2_pts  = (float*)alloc((size_t)B * S2 * 256 * 4);
    _Float16* Y0      = (_Float16*)alloc((size_t)M1 * 128 * 2);  // 64 MB
    _Float16* Y1      = (_Float16*)alloc((size_t)M1 * 128 * 2);  // 64 MB
    float*    sumB    = (float*)alloc(1024 * 4);
    float*    sqB     = (float*)alloc(1024 * 4);
    float*    scaleB  = (float*)alloc(1024 * 4);
    float*    shiftB  = (float*)alloc(1024 * 4);

    // MLP driver: 3 layers of GEMM -> stats -> BN finalize.
    // Layer 0 of each stage: ragged K, no input BN  -> gemm<false>.
    // Layers 1,2: K%32==0, fused BN+ReLU on A       -> gemm<true>.
    auto run_mlp = [&](const _Float16* A0, int M, int stage) -> const _Float16* {
        const _Float16* curA = A0;
        _Float16* bufs[2] = {Y0, Y1};
        int cur = 0;
        for (int l = 0; l < 3; ++l) {
            const int li = stage * 3 + l;
            const int K = cins[li], Nn = couts[li];
            _Float16* out = bufs[cur];
            dim3 grid(M / 128, Nn / 64);
            if (l == 0)
                gemm_wmma_kernel<false><<<grid, 256, 0, stream>>>(
                    curA, M, K, nullptr, nullptr, Wp[li], bp[li], Nn, out);
            else
                gemm_wmma_kernel<true><<<grid, 256, 0, stream>>>(
                    curA, M, K, scaleB, shiftB, Wp[li], bp[li], Nn, out);
            channel_stats_kernel<<<Nn, 256, 0, stream>>>(out, M, Nn, sumB, sqB);
            bn_finalize_kernel<<<(Nn + 255) / 256, 256, 0, stream>>>(
                sumB, sqB, gp[li], betap[li], 1.0f / (float)M, Nn, scaleB, shiftB);
            curA = out;
            cur ^= 1;
        }
        return curA;
    };

    // ---------------- stage sa1 ----------------
    fps_kernel<<<B, 512, 0, stream>>>(points, N1, 4, S1, idx1);
    gather_xyz_kernel<<<(B * S1 + 255) / 256, 256, 0, stream>>>(
        points, N1, 4, idx1, S1, B * S1, l1_xyz);
    ball_query_kernel<<<(B * S1 * 32) / 256, 256, 0, stream>>>(
        points, N1, 4, l1_xyz, S1, 0.01f, NS1, ball1);
    group1_kernel<<<(M1 + 255) / 256, 256, 0, stream>>>(
        points, ball1, l1_xyz, A1, M1, S1, NS1, N1);
    {
        const _Float16* Ylast = run_mlp(A1, M1, 0);
        bn_relu_maxpool_kernel<<<(B * S1 * 128 + 255) / 256, 256, 0, stream>>>(
            Ylast, scaleB, shiftB, B * S1, NS1, 128, l1_pts);
    }

    // ---------------- stage sa2 ----------------
    fps_kernel<<<B, 512, 0, stream>>>(l1_xyz, S1, 3, S2, idx2);
    gather_xyz_kernel<<<(B * S2 + 255) / 256, 256, 0, stream>>>(
        l1_xyz, S1, 3, idx2, S2, B * S2, l2_xyz);
    ball_query_kernel<<<(B * S2 * 32) / 256, 256, 0, stream>>>(
        l1_xyz, S1, 3, l2_xyz, S2, 0.09f, NS2, ball2);
    group2_kernel<<<((M2 * 131) + 255) / 256, 256, 0, stream>>>(
        l1_xyz, l1_pts, l2_xyz, ball2, A2, M2 * 131);
    {
        const _Float16* Ylast = run_mlp(A2, M2, 1);
        bn_relu_maxpool_kernel<<<(B * S2 * 256 + 255) / 256, 256, 0, stream>>>(
            Ylast, scaleB, shiftB, B * S2, NS2, 256, l2_pts);
    }

    // ---------------- stage sa3 (group-all) ----------------
    group3_kernel<<<((M3 * 259) + 255) / 256, 256, 0, stream>>>(
        l2_xyz, l2_pts, A3, M3 * 259);
    {
        const _Float16* Ylast = run_mlp(A3, M3, 2);
        bn_relu_maxpool_kernel<<<(B * 1024 + 255) / 256, 256, 0, stream>>>(
            Ylast, scaleB, shiftB, B, K3, 1024, (float*)d_out);
    }
}
